// Topological_Loss_55396488184223
// MI455X (gfx1250) — matrix operand order repaired and checked by reading, not compiled, not run on hardware
//
#include <hip/hip_runtime.h>
#include <math.h>

typedef __attribute__((ext_vector_type(16))) _Float16 v16h;
typedef __attribute__((ext_vector_type(8)))  _Float16 h8;
typedef __attribute__((ext_vector_type(4)))  _Float16 h4;
typedef __attribute__((ext_vector_type(8)))  float    v8f;

#define N 512            // points per cloud == dims per point
#define NMAT 128         // 64 samples * {P, M}
#define MATELTS (512*512)
#define FLT_BIG 3.402823466e38f

// fast sigmoid: v_exp + v_add + v_rcp (avoids IEEE division expansion)
__device__ __forceinline__ float fsig(float x) {
    return __builtin_amdgcn_rcpf(1.f + __expf(-x));
}

// ---------------- init: zero per-matrix max accumulators ----------------
__global__ void k_init(unsigned* maxv) {
    int t = threadIdx.x;
    if (t < NMAT) maxv[t] = 0u;
}

// ---------------- prep: f16 copy (sigmoid for P) + row squared norms ----------------
// one wave per row; float4 loads, b64 f16 stores, fp32 norm of the f16-rounded values
__global__ void k_prep(const float* __restrict__ mo, const float* __restrict__ lb,
                       _Float16* __restrict__ X16, float* __restrict__ sq) {
    int lane = threadIdx.x;                    // 0..31
    int rg   = blockIdx.x * 8 + threadIdx.y;   // global row id 0..65535
    int m = rg >> 9, i = rg & 511;
    int b = m >> 1, which = m & 1;
    const float* src = (which ? lb : mo) + (size_t)b * MATELTS + (size_t)i * N;
    _Float16* dst = X16 + (size_t)m * MATELTS + (size_t)i * N;
    float s = 0.f;
#pragma unroll
    for (int t = 0; t < 4; ++t) {
        int o = (lane << 2) + (t << 7);
        float4 x = *(const float4*)(src + o);
        if (!which) { x.x = fsig(x.x); x.y = fsig(x.y); x.z = fsig(x.z); x.w = fsig(x.w); }
        h4 hx = { (_Float16)x.x, (_Float16)x.y, (_Float16)x.z, (_Float16)x.w };
        *(h4*)(dst + o) = hx;
        float a = (float)hx[0], c = (float)hx[1], d = (float)hx[2], e = (float)hx[3];
        s += a * a + c * c + d * d + e * e;
    }
#pragma unroll
    for (int off = 16; off > 0; off >>= 1) s += __shfl_xor(s, off, 32);
    if (lane == 0) sq[(size_t)m * N + i] = s;
}

// ---------------- distance epilogue shared by both gram variants ----------------
__device__ __forceinline__ void dist_epilogue(v8f acc, int m, int i0, int j0,
                                              int mrow, int hi, int lane,
                                              const float* __restrict__ sq,
                                              float* __restrict__ dist,
                                              unsigned* __restrict__ maxv) {
    // C/D layout: lanes 0-15: (M=r, N=lane); lanes 16-31: (M=r+8, N=lane-16)
    const float* sqm = sq + (size_t)m * N;
    float* dm = dist + (size_t)m * MATELTS;
    int j = j0 + mrow;
    float sqj = sqm[j];
    float lmax = 0.f;
#pragma unroll
    for (int r = 0; r < 8; ++r) {
        int i = i0 + 8 * hi + r;
        float d2 = sqm[i] + sqj - 2.f * acc[r];
        float d = sqrtf(fmaxf(d2, 0.f));
        dm[(size_t)i * N + j] = d;
        lmax = fmaxf(lmax, d);
    }
#pragma unroll
    for (int off = 16; off > 0; off >>= 1) lmax = fmaxf(lmax, __shfl_xor(lmax, off, 32));
    if (lane == 0) atomicMax(&maxv[m], __float_as_uint(lmax));  // d>=0: uint max == float max
}

// ---------------- FAST gram: direct-from-memory WMMA fragments, no LDS ----------------
// A frag (16x32 f16, contiguous k order): lane=row i0+(lane&15); halves 0-7  = K kk+hi*16..+7
//                                                        halves 8-15 = K kk+hi*16+8..+15
// B frag (32x16 f16):             lane=col j0+(lane&15); halves 0-15 = K kk+hi*16..+15
// For X·X^T both fragments are contiguous row segments of row-major f16 X.
__global__ void k_gram_fast(const _Float16* __restrict__ X16, const float* __restrict__ sq,
                            float* __restrict__ dist, unsigned* __restrict__ maxv) {
    int lane = threadIdx.x;
    int m    = blockIdx.y;
    int tile = blockIdx.x * 8 + threadIdx.y;   // 0..1023
    int i0 = (tile >> 5) * 16, j0 = (tile & 31) * 16;
    int mrow = lane & 15, hi = lane >> 4;

    const _Float16* arow = X16 + (size_t)m * MATELTS + (size_t)(i0 + mrow) * N;
    const _Float16* brow = X16 + (size_t)m * MATELTS + (size_t)(j0 + mrow) * N;

    v8f acc = {};
#pragma unroll 4
    for (int kk = 0; kk < N; kk += 32) {
        h8 alo = *(const h8*)(arow + kk + hi * 16);       // 16B
        h8 ahi = *(const h8*)(arow + kk + hi * 16 + 8);   // 16B
        v16h av = __builtin_shufflevector(alo, ahi, 0, 1, 2, 3, 4, 5, 6, 7,
                                          8, 9, 10, 11, 12, 13, 14, 15);
        v16h bv = *(const v16h*)(brow + kk + hi * 16);    // 32B
        acc = __builtin_amdgcn_wmma_f32_16x16x32_f16(
            false, av, false, bv, (short)0, acc, false, false);
    }
    dist_epilogue(acc, m, i0, j0, mrow, hi, lane, sq, dist, maxv);
}

// ---------------- FALLBACK gram (small ws): stage fp32->f16 via LDS ----------------
__global__ void k_gram_lds(const float* __restrict__ mo, const float* __restrict__ lb,
                           const float* __restrict__ sq, float* __restrict__ dist,
                           unsigned* __restrict__ maxv) {
    __shared__ _Float16 As[8][16 * 32];
    __shared__ _Float16 Bs[8][16 * 32];

    int lane = threadIdx.x, w = threadIdx.y, m = blockIdx.y;
    int b = m >> 1, which = m & 1;
    int tile = blockIdx.x * 8 + w;
    int i0 = (tile >> 5) * 16, j0 = (tile & 31) * 16;
    const float* src = (which ? lb : mo) + (size_t)b * MATELTS;

    int mrow = lane & 15, hi = lane >> 4;
    int lr = lane >> 3, lc = (lane & 7) << 2;

    v8f acc = {};
    for (int kk = 0; kk < N; kk += 32) {
#pragma unroll
        for (int t = 0; t < 4; ++t) {
            int r = (t << 2) + lr;
            float4 va = *(const float4*)(src + (size_t)(i0 + r) * N + kk + lc);
            float4 vb = *(const float4*)(src + (size_t)(j0 + r) * N + kk + lc);
            if (!which) {
                va.x = fsig(va.x); va.y = fsig(va.y); va.z = fsig(va.z); va.w = fsig(va.w);
                vb.x = fsig(vb.x); vb.y = fsig(vb.y); vb.z = fsig(vb.z); vb.w = fsig(vb.w);
            }
            h4 ha = { (_Float16)va.x, (_Float16)va.y, (_Float16)va.z, (_Float16)va.w };
            h4 hb = { (_Float16)vb.x, (_Float16)vb.y, (_Float16)vb.z, (_Float16)vb.w };
            *(h4*)&As[w][r * 32 + lc] = ha;
            *(h4*)&Bs[w][r * 32 + lc] = hb;
        }
        v16h av, bv;
#pragma unroll
        for (int v = 0; v < 8; ++v) {
#pragma unroll
            for (int h = 0; h < 2; ++h) {
                int ka = hi * 16 + v * 2 + h;
                int kb = hi * 16 + v * 2 + h;
                av[v * 2 + h] = As[w][mrow * 32 + ka];
                bv[v * 2 + h] = Bs[w][mrow * 32 + kb];
            }
        }
        acc = __builtin_amdgcn_wmma_f32_16x16x32_f16(
            false, av, false, bv, (short)0, acc, false, false);
    }
    dist_epilogue(acc, m, i0, j0, mrow, hi, lane, sq, dist, maxv);
}

// ---------------- row norms only (fallback path) ----------------
__global__ void k_rownorm(const float* __restrict__ mo, const float* __restrict__ lb,
                          float* __restrict__ sq) {
    int lane = threadIdx.x;
    int rg   = blockIdx.x * 8 + threadIdx.y;
    int m = rg >> 9, i = rg & 511;
    int b = m >> 1, which = m & 1;
    const float* src = (which ? lb : mo) + (size_t)b * MATELTS + (size_t)i * N;
    float s = 0.f;
#pragma unroll
    for (int t = 0; t < 4; ++t) {
        float4 x = *(const float4*)(src + (lane << 2) + (t << 7));
        if (!which) { x.x = fsig(x.x); x.y = fsig(x.y); x.z = fsig(x.z); x.w = fsig(x.w); }
        float a = (float)(_Float16)x.x, c = (float)(_Float16)x.y;
        float d = (float)(_Float16)x.z, e = (float)(_Float16)x.w;
        s += a * a + c * c + d * d + e * e;
    }
#pragma unroll
    for (int off = 16; off > 0; off >>= 1) s += __shfl_xor(s, off, 32);
    if (lane == 0) sq[(size_t)m * N + i] = s;
}

// ---------------- Prim's MST + signature accumulation ----------------
__global__ void k_mst(const float* __restrict__ dist, const unsigned* __restrict__ maxv,
                      float* __restrict__ partial) {
    __shared__ float swv[16];
    __shared__ int   swj[16];
    __shared__ int   sws[16];
    __shared__ int   bj, bsrc;
    __shared__ float bval;

    int m = blockIdx.x;
    int other = m ^ 1;
    const float* dA = dist + (size_t)m * MATELTS;
    const float* dB = dist + (size_t)other * MATELTS;
    float invA = 1.f / __uint_as_float(maxv[m]);
    float invB = 1.f / __uint_as_float(maxv[other]);

    int j = threadIdx.x;                 // 0..511
    bool inTree = (j == 0);
    float mind = dA[j];
    int   msrc = 0;
    float s = 0.f;

    for (int it = 0; it < N - 1; ++it) {
        float v = inTree ? FLT_BIG : mind;
        int   idx = j, srcv = msrc;
#pragma unroll
        for (int off = 16; off > 0; off >>= 1) {
            float ov = __shfl_xor(v, off, 32);
            int   oi = __shfl_xor(idx, off, 32);
            int   os = __shfl_xor(srcv, off, 32);
            if (ov < v || (ov == v && oi < idx)) { v = ov; idx = oi; srcv = os; }
        }
        if ((j & 31) == 0) { int wv = j >> 5; swv[wv] = v; swj[wv] = idx; sws[wv] = srcv; }
        __syncthreads();
        if (j < 32) {
            float v2 = (j < 16) ? swv[j] : FLT_BIG;
            int   i2 = (j < 16) ? swj[j] : 0x7fffffff;
            int   s2 = (j < 16) ? sws[j] : 0;
#pragma unroll
            for (int off = 8; off > 0; off >>= 1) {
                float ov = __shfl_xor(v2, off, 32);
                int   oi = __shfl_xor(i2, off, 32);
                int   os = __shfl_xor(s2, off, 32);
                if (ov < v2 || (ov == v2 && oi < i2)) { v2 = ov; i2 = oi; s2 = os; }
            }
            if (j == 0) { bj = i2; bsrc = s2; bval = v2; }
        }
        __syncthreads();
        int jstar = bj, srcstar = bsrc;
        if (j == 0) {
            float da = bval;                              // dA[srcstar][jstar]
            float db = dB[(size_t)srcstar * N + jstar];
            float diff = da * invA - db * invB;
            s += diff * diff;
        }
        if (j == jstar) inTree = true;
        float rowv = dA[(size_t)jstar * N + j];
        if (!inTree && rowv < mind) { mind = rowv; msrc = jstar; }
    }
    if (j == 0) partial[m] = s;
}

// ---------------- deterministic final reduction ----------------
__global__ void k_final(const float* __restrict__ partial, float* __restrict__ out) {
    __shared__ float sh[NMAT];
    int t = threadIdx.x;
    sh[t] = partial[t];
    __syncthreads();
    for (int off = 64; off > 0; off >>= 1) {
        if (t < off) sh[t] += sh[t + off];
        __syncthreads();
    }
    // LAM * (1/B) * 0.5 = 0.5 * (1/64) * 0.5
    if (t == 0) out[0] = 0.00390625f * sh[0];
}

extern "C" void kernel_launch(void* const* d_in, const int* in_sizes, int n_in,
                              void* d_out, int out_size, void* d_ws, size_t ws_size,
                              hipStream_t stream) {
    const float* mo = (const float*)d_in[0];   // model_output [64,1,512,512] f32
    const float* lb = (const float*)d_in[1];   // labels       [64,1,512,512] f32
    float* out = (float*)d_out;

    char* ws = (char*)d_ws;
    size_t distB = (size_t)NMAT * MATELTS * 4;            // 128 MiB
    size_t x16B  = (size_t)NMAT * MATELTS * 2;            //  64 MiB
    size_t sqB   = (size_t)NMAT * N * 4;                  // 256 KiB

    float* dist = (float*)ws;
    // small-ws fallback layout omits X16
    bool fast = (ws_size >= distB + x16B + sqB + 1024);

    if (fast) {
        _Float16* X16     = (_Float16*)(ws + distB);
        float*    sq      = (float*)(ws + distB + x16B);
        unsigned* maxv    = (unsigned*)((char*)sq + sqB);
        float*    partial = (float*)((char*)maxv + 512);

        k_init<<<1, 128, 0, stream>>>(maxv);
        k_prep<<<dim3(8192), dim3(32, 8), 0, stream>>>(mo, lb, X16, sq);
        k_gram_fast<<<dim3(128, 128), dim3(32, 8), 0, stream>>>(X16, sq, dist, maxv);
        k_mst<<<dim3(NMAT), dim3(N), 0, stream>>>(dist, maxv, partial);
        k_final<<<1, 128, 0, stream>>>(partial, out);
    } else {
        float*    sq      = (float*)(ws + distB);
        unsigned* maxv    = (unsigned*)((char*)sq + sqB);
        float*    partial = (float*)((char*)maxv + 512);

        k_init<<<1, 128, 0, stream>>>(maxv);
        k_rownorm<<<dim3(8192), dim3(32, 8), 0, stream>>>(mo, lb, sq);
        k_gram_lds<<<dim3(128, 128), dim3(32, 8), 0, stream>>>(mo, lb, sq, dist, maxv);
        k_mst<<<dim3(NMAT), dim3(N), 0, stream>>>(dist, maxv, partial);
        k_final<<<1, 128, 0, stream>>>(partial, out);
    }
}
